// AggregateModel_41059887349975
// MI455X (gfx1250) — matrix operand order repaired and checked by reading, not compiled, not run on hardware
//
#include <hip/hip_runtime.h>
#include <hip/hip_bf16.h>
#include <math.h>

// ---------- problem constants (match reference) ----------
#define DDIM   2048
#define BINS   128
#define NOUT   129          // BINS + y_extra
#define NPAD   144          // 9 * 16
#define NT     9            // n-tiles of 16
#define KTILE  32
#define MTILE  64           // rows per block (4 waves x 16)
#define ITERS  112          // BINS - N
#define AVAL   0.0625f
#define TINYV  1e-30f

typedef __attribute__((ext_vector_type(16))) __bf16 v16bf;
typedef __attribute__((ext_vector_type(8)))  __bf16 v8bf;
typedef __attribute__((ext_vector_type(8)))  float  v8f;
typedef __attribute__((ext_vector_type(4)))  float  v4f;

#if __has_builtin(__builtin_amdgcn_tensor_load_to_lds)
#define USE_TDM 1
typedef unsigned int u32x4 __attribute__((ext_vector_type(4)));
typedef int          i32x8 __attribute__((ext_vector_type(8)));
typedef int          i32x4 __attribute__((ext_vector_type(4)));
#endif

__device__ __forceinline__ unsigned lds_offset_of(const void* p) {
    // generic -> LDS(AS3) -> 32-bit LDS byte offset
    return (unsigned)(unsigned long long)(__attribute__((address_space(3))) const void*)p;
}

// =====================================================================
// Kernel 1: y = x@W + b, softmax over first 128 cols, fused epilogue.
// A (the 134MB HBM stream) is staged via the Tensor Data Mover;
// B is converted to bf16 on the vector path (L2-resident, ~1MB).
// =====================================================================
__global__ __launch_bounds__(128) void gemm_softmax_kernel(
    const float* __restrict__ x, const float* __restrict__ W,
    const float* __restrict__ bias, float* __restrict__ out)
{
    __shared__ float  ldsAf32[MTILE][KTILE];   // 64x32 f32 (8 KB), filled by TDM
    __shared__ __bf16 ldsBt[NPAD][KTILE];      // 144x32 bf16 (9 KB), B transposed
    __shared__ float  ldsBias[NOUT];

    const int t    = threadIdx.x;
    const int lane = t & 31;
    const int wv   = t >> 5;              // wave 0..3 -> 16-row strip
    const int half = lane >> 4;           // 0 | 1
    const int l16  = lane & 15;
    const int mbase = blockIdx.x * MTILE;

    if (t < NOUT) ldsBias[t] = bias[t];

#if USE_TDM
    const unsigned ldsA_off = lds_offset_of(&ldsAf32[0][0]);
    const unsigned rows     = (unsigned)gridDim.x * MTILE;
#endif

    v8f acc[NT];
    #pragma unroll
    for (int i = 0; i < NT; ++i)
        #pragma unroll
        for (int j = 0; j < 8; ++j) acc[i][j] = 0.0f;

    for (int kt = 0; kt < DDIM; kt += KTILE) {
        __syncthreads();                  // WAR guard on LDS tiles

#if USE_TDM
        // ---- TDM: 2D tile (64 rows x 32 cols, 4B elems) of x -> LDS ----
        if (wv == 0) {
            const unsigned long long ga =
                (unsigned long long)(size_t)x + ((size_t)mbase * DDIM + kt) * 4ull;
            u32x4 g0;
            g0[0] = 1u;                                   // count=1, user mode
            g0[1] = ldsA_off;                             // lds_addr
            g0[2] = (unsigned)ga;                         // global_addr[31:0]
            g0[3] = (unsigned)((ga >> 32) & 0x01FFFFFFu)  // global_addr[56:32]
                  | 0x80000000u;                          // type = 2 ("image")
            i32x8 g1;
            g1[0] = (int)(2u << 16);                      // data_size = 4B
            g1[1] = (int)((DDIM & 0xFFFFu) << 16);        // tensor_dim0[15:0]
            g1[2] = (int)((DDIM >> 16) | ((rows & 0xFFFFu) << 16)); // td0 hi | td1 lo
            g1[3] = (int)((rows >> 16) | (32u << 16));    // td1 hi | tile_dim0=32
            g1[4] = 64;                                   // tile_dim1=64, tile_dim2=0
            g1[5] = DDIM;                                 // tensor_dim0_stride
            g1[6] = 0;
            g1[7] = 0;
            i32x4 gz4; gz4[0] = 0; gz4[1] = 0; gz4[2] = 0; gz4[3] = 0;  // groups 2/3: 2D tile
            i32x8 gz8;
            #pragma unroll
            for (int i = 0; i < 8; ++i) gz8[i] = 0;
            __builtin_amdgcn_tensor_load_to_lds(g0, g1, gz4, gz4, gz8, 0);
        }
#else
        // ---- fallback: cooperative vector loads of the A tile ----
        {
            const int row = t >> 1;
            const int k0  = (t & 1) * 16;
            const float* src = x + (size_t)(mbase + row) * DDIM + kt + k0;
            v4f f0 = *(const v4f*)(src + 0);
            v4f f1 = *(const v4f*)(src + 4);
            v4f f2 = *(const v4f*)(src + 8);
            v4f f3 = *(const v4f*)(src + 12);
            *(v4f*)&ldsAf32[row][k0 + 0]  = f0;
            *(v4f*)&ldsAf32[row][k0 + 4]  = f1;
            *(v4f*)&ldsAf32[row][k0 + 8]  = f2;
            *(v4f*)&ldsAf32[row][k0 + 12] = f3;
        }
#endif
        // ---- stage B tile transposed (144 cols x 32 k), zero-pad n>=129 ----
        if (t == 0 && kt + KTILE < DDIM)
            __builtin_prefetch(&W[(size_t)(kt + KTILE) * NOUT], 0, 1);  // global_prefetch_b8
        for (int idx = t; idx < KTILE * NPAD; idx += 128) {
            const int k = idx / NPAD;
            const int n = idx - k * NPAD;
            const float v = (n < NOUT) ? W[(size_t)(kt + k) * NOUT + n] : 0.0f;
            ldsBt[n][k] = (__bf16)v;
        }
#if USE_TDM
        if (wv == 0) __builtin_amdgcn_s_wait_tensorcnt(0);
#endif
        __syncthreads();

        // ---- A fragment: 16-bit 16x32 layout ----
        // lane<16: K 0..7 (elems 0..7), K 16..23 (elems 8..15); lane>=16: +8
        v16bf afrag;
        {
            const int r  = wv * 16 + l16;
            const int k0 = half * 8;
            v4f a0 = *(const v4f*)&ldsAf32[r][k0 + 0];
            v4f a1 = *(const v4f*)&ldsAf32[r][k0 + 4];
            v4f a2 = *(const v4f*)&ldsAf32[r][k0 + 16];
            v4f a3 = *(const v4f*)&ldsAf32[r][k0 + 20];
            #pragma unroll
            for (int i = 0; i < 4; ++i) {
                afrag[i]      = (__bf16)a0[i];
                afrag[4 + i]  = (__bf16)a1[i];
                afrag[8 + i]  = (__bf16)a2[i];
                afrag[12 + i] = (__bf16)a3[i];
            }
        }
        // ---- load all 9 B fragments first (overlap ds latency), then WMMAs ----
        // B 32x16 16-bit layout: lane<16 K0..7 (elems 0..7), K16..23 (elems 8..15); lane>=16: +8; col=lane%16
        v16bf bfrag[NT];
        #pragma unroll
        for (int nt = 0; nt < NT; ++nt) {
            const int n  = nt * 16 + l16;
            const int kb = half * 8;
            v8bf b0 = *(const v8bf*)(&ldsBt[n][kb]);
            v8bf b1 = *(const v8bf*)(&ldsBt[n][kb + 16]);
            #pragma unroll
            for (int i = 0; i < 8; ++i) { bfrag[nt][i] = b0[i]; bfrag[nt][8 + i] = b1[i]; }
        }
        #pragma unroll
        for (int nt = 0; nt < NT; ++nt)
            acc[nt] = __builtin_amdgcn_wmma_f32_16x16x32_bf16(
                false, afrag, false, bfrag[nt], (short)0, acc[nt], false, false);
    }

    // ---- fused bias + softmax epilogue ----
    // C/D layout: elem j of lane l  ->  row = j + (l>=16 ? 8 : 0), col = nt*16 + l%16
    #pragma unroll
    for (int j = 0; j < 8; ++j) {
        const int rl  = j + half * 8;
        const int row = mbase + wv * 16 + rl;
        float v[8];
        #pragma unroll
        for (int nt = 0; nt < 8; ++nt) v[nt] = acc[nt][j] + ldsBias[nt * 16 + l16];

        float mx = v[0];
        #pragma unroll
        for (int nt = 1; nt < 8; ++nt) mx = fmaxf(mx, v[nt]);
        #pragma unroll
        for (int m = 1; m < 16; m <<= 1) mx = fmaxf(mx, __shfl_xor(mx, m, 32));

        float se = 0.0f;
        float e[8];
        #pragma unroll
        for (int nt = 0; nt < 8; ++nt) { e[nt] = __expf(v[nt] - mx); se += e[nt]; }
        #pragma unroll
        for (int m = 1; m < 16; m <<= 1) se += __shfl_xor(se, m, 32);
        const float inv = 1.0f / se;

        float* orow = out + (size_t)row * NOUT;
        #pragma unroll
        for (int nt = 0; nt < 8; ++nt) orow[nt * 16 + l16] = e[nt] * inv;
        if (l16 == 0) orow[BINS] = acc[8][j] + ldsBias[BINS];   // y_extra column
    }
}

// =====================================================================
// Kernel 2: iterative bin merge, one wave32 per row, in-place on d_out.
// =====================================================================
__device__ __forceinline__ float rng_uniform(unsigned it, unsigned row, unsigned c) {
    unsigned long long z = ((unsigned long long)it << 40) ^
                           ((unsigned long long)row << 8) ^ (unsigned long long)c;
    z += 0x9E3779B97F4A7C15ULL;
    z = (z ^ (z >> 30)) * 0xBF58476D1CE4E5B9ULL;
    z = (z ^ (z >> 27)) * 0x94D049BB133111EBULL;
    z ^= z >> 31;
    return (float)(z >> 40) * (1.0f / 16777216.0f);   // [0,1)
}

// CDF-inversion categorical draw over 128 per-wave weights (4 per lane).
__device__ __forceinline__ int sample_categorical(const float w[4], int lane, float u) {
    float ls[4];
    ls[0] = w[0]; ls[1] = ls[0] + w[1]; ls[2] = ls[1] + w[2]; ls[3] = ls[2] + w[3];
    const float tl = ls[3];
    float run = tl;                                   // inclusive scan of lane totals
    #pragma unroll
    for (int ofs = 1; ofs < 32; ofs <<= 1) {
        float o = __shfl_up(run, ofs, 32);
        if (lane >= ofs) run += o;
    }
    const float total = __shfl(run, 31, 32);
    if (!(total > 0.0f)) return -1;
    const float ex = run - tl;                        // exclusive prefix
    const float target = u * total;
    int cand = 128, lastnz = -1;
    #pragma unroll
    for (int s = 0; s < 4; ++s) {
        if (w[s] > 0.0f) {
            lastnz = lane * 4 + s;
            if (cand == 128 && (ex + ls[s]) > target) cand = lane * 4 + s;
        }
    }
    #pragma unroll
    for (int m = 16; m >= 1; m >>= 1) cand = min(cand, __shfl_xor(cand, m, 32));
    #pragma unroll
    for (int m = 16; m >= 1; m >>= 1) lastnz = max(lastnz, __shfl_xor(lastnz, m, 32));
    if (cand == 128) cand = lastnz;                   // fp edge fallback
    return cand;
}

__global__ __launch_bounds__(256) void aggregate_kernel(
    const float* __restrict__ bounds, float* __restrict__ out)
{
    __shared__ float raw[BINS - 1];   // 127 sorted bounds
    __shared__ float mid[BINS];       // 128 midpoints

    const int t = threadIdx.x, lane = t & 31, wv = t >> 5;
    if (t < BINS - 1) raw[t] = bounds[t];
    __syncthreads();
    if (t == 0) {
        const float eps = (raw[1] - raw[0] + raw[126] - raw[125]) * 0.5f;
        mid[0] = raw[0] - eps * 0.5f;
        for (int i = 1; i < 127; ++i) mid[i] = (raw[i - 1] + raw[i]) * 0.5f;
        mid[127] = raw[126] + eps * 0.5f;
    }
    __syncthreads();

    const int row = blockIdx.x * 8 + wv;
    float* xr = out + (size_t)row * NOUT;
    float xv[4];
    #pragma unroll
    for (int s = 0; s < 4; ++s) xv[s] = xr[lane * 4 + s];

    for (int it = 0; it < ITERS; ++it) {
        // ---- p = sum of top-16 bin masses (argmax-and-remove x16) ----
        float tc[4];
        #pragma unroll
        for (int s = 0; s < 4; ++s) tc[s] = xv[s];
        float p = 0.0f;
        for (int i = 0; i < 16; ++i) {
            float bv = -1.0f; int bi = 0;
            #pragma unroll
            for (int s = 0; s < 4; ++s)
                if (tc[s] > bv) { bv = tc[s]; bi = lane * 4 + s; }
            #pragma unroll
            for (int m = 16; m >= 1; m >>= 1) {
                float ov = __shfl_xor(bv, m, 32); int oi = __shfl_xor(bi, m, 32);
                if (ov > bv || (ov == bv && oi < bi)) { bv = ov; bi = oi; }
            }
            p += bv;
            if ((bi >> 2) == lane) tc[bi & 3] = -1.0f;
        }
        if (!(p < 1.0f)) break;   // mask false -> row frozen forever (monotone)

        // ---- draw idx1 ~ categorical(x * [x < a]) ----
        float w[4];
        #pragma unroll
        for (int s = 0; s < 4; ++s) w[s] = (xv[s] < AVAL) ? xv[s] : 0.0f;
        const int idx1 = sample_categorical(w, lane, rng_uniform(it, row, 0));
        if (idx1 < 0) continue;
        const float b1 = mid[idx1];
        const float x1 = __shfl(xv[idx1 & 3], idx1 >> 2, 32);

        // ---- idx2 ~ categorical((x/|mid-b1|) * [x < a]) ----
        #pragma unroll
        for (int s = 0; s < 4; ++s) {
            const float d  = fabsf(mid[lane * 4 + s] - b1);
            const float ww = (d > 0.0f) ? xv[s] / d : 0.0f;
            w[s] = (xv[s] < AVAL) ? ww : 0.0f;
        }
        const int idx2 = sample_categorical(w, lane, rng_uniform(it, row, 1));
        if (idx2 < 0) continue;
        const float b2 = mid[idx2];
        const float x2 = __shfl(xv[idx2 & 3], idx2 >> 2, 32);

        // ---- merge: zero both, deposit mass at searchsorted(v, raw) ----
        if ((idx1 >> 2) == lane) xv[idx1 & 3] = 0.0f;
        if ((idx2 >> 2) == lane) xv[idx2 & 3] = 0.0f;
        const float mass = x1 + x2;
        const float v = (b1 * x1 + b2 * x2) / fmaxf(mass, TINYV);
        int cnt = 0;
        #pragma unroll
        for (int s = 0; s < 4; ++s) {
            const int jj = lane * 4 + s;
            if (jj < BINS - 1 && v >= raw[jj]) ++cnt;
        }
        #pragma unroll
        for (int m = 16; m >= 1; m >>= 1) cnt += __shfl_xor(cnt, m, 32);
        if ((cnt >> 2) == lane) xv[cnt & 3] += mass;
    }

    #pragma unroll
    for (int s = 0; s < 4; ++s) xr[lane * 4 + s] = xv[s];
}

// =====================================================================
extern "C" void kernel_launch(void* const* d_in, const int* in_sizes, int n_in,
                              void* d_out, int out_size, void* d_ws, size_t ws_size,
                              hipStream_t stream) {
    (void)n_in; (void)out_size; (void)d_ws; (void)ws_size;
    const float* x      = (const float*)d_in[0];   // [rows, 2048]
    const float* W      = (const float*)d_in[1];   // [2048, 129]
    const float* b      = (const float*)d_in[2];   // [129]
    const float* bounds = (const float*)d_in[3];   // [127]
    float* out = (float*)d_out;                    // [rows, 129]

    const int rows = in_sizes[0] / DDIM;           // 16384

    gemm_softmax_kernel<<<rows / MTILE, 128, 0, stream>>>(x, W, b, out);
    aggregate_kernel<<<rows / 8, 256, 0, stream>>>(bounds, out);
}
